// EncoderLayer_57475252355154
// MI455X (gfx1250) — matrix operand order repaired and checked by reading, not compiled, not run on hardware
//
#include <hip/hip_runtime.h>
#include <hip/hip_bf16.h>
#include <math.h>

// ---------------------------------------------------------------------------
// CDNA5 (gfx1250) encoder layer: bf16 WMMA GEMMs + fused flash attention,
// TDM (tensor_load_to_lds) staging for attention K/V tiles when available.
// Wave32; WMMA shape 16x16x32 bf16 -> f32 accumulate.
// ---------------------------------------------------------------------------

typedef __bf16 bf16_t;
typedef __attribute__((ext_vector_type(16))) __bf16 v16bf;
typedef __attribute__((ext_vector_type(8)))  __bf16 v8bf;
typedef __attribute__((ext_vector_type(4)))  __bf16 v4bf;
typedef __attribute__((ext_vector_type(2)))  __bf16 v2bf;
typedef __attribute__((ext_vector_type(8)))  float  v8f;
typedef __attribute__((ext_vector_type(4)))  unsigned int v4u;
typedef __attribute__((ext_vector_type(8)))  int v8i;
typedef __attribute__((ext_vector_type(4)))  int v4i;

union V16 { v16bf v; v8bf h[2]; };

// Model dims (compile-time).
constexpr int cB = 2, cS = 2048, cD = 1024, cH = 16, cDH = 64, cF = 4096;
constexpr int cBS = cB * cS;  // 4096

__device__ __forceinline__ v8f wmma_bf16(v16bf a, v16bf b, v8f c) {
  return __builtin_amdgcn_wmma_f32_16x16x32_bf16(
      false, a, false, b, (short)0, c, false, false);
}

__device__ __forceinline__ float gelu_exact(float v) {
  return 0.5f * v * (1.0f + erff(v * 0.70710678118654752f));
}

// ---------------------------------------------------------------------------
// TDM: 2D tile load global -> LDS via Tensor Data Mover (ISA ch. 8 D#).
// data_size = 2 bytes (bf16). Issued by one wave; tracked with TENSORcnt.
// This toolchain exposes the 6-arg builtin:
//   (uint32x4 g0, int32x8 g1, int32x4, int32x4, int32x8, i32 cpol)
// ---------------------------------------------------------------------------
#if __has_builtin(__builtin_amdgcn_tensor_load_to_lds)
#define USE_TDM 1
#else
#define USE_TDM 0
#endif

#if USE_TDM
__device__ __forceinline__ void tdm_load_2d_bf16(
    unsigned lds_off, const void* gptr,
    unsigned tensor_d0, unsigned tensor_d1,
    unsigned tile_d0, unsigned tile_d1,
    unsigned long long stride0_elems) {
  unsigned long long ga = (unsigned long long)gptr;
  v4u g0;
  g0[0] = 1u;                                         // count=1, user D#
  g0[1] = lds_off;                                    // LDS byte address
  g0[2] = (unsigned)(ga & 0xFFFFFFFFull);             // global_addr[31:0]
  g0[3] = (unsigned)((ga >> 32) & 0x1FFFFFFull)       // global_addr[56:32]
          | (2u << 30);                               // type=2 ("image")
  v8i g1;
  g1[0] = (int)(1u << 16);                            // wg_mask=0, data_size=1 (2B)
  g1[1] = (int)((tensor_d0 & 0xFFFFu) << 16);         // tensor_dim0[15:0] @63:48
  g1[2] = (int)((tensor_d0 >> 16) |                   // tensor_dim0[31:16]
                ((tensor_d1 & 0xFFFFu) << 16));       // tensor_dim1[15:0]
  g1[3] = (int)((tensor_d1 >> 16) |                   // tensor_dim1[31:16]
                (tile_d0 << 16));                     // tile_dim0 @127:112
  g1[4] = (int)(tile_d1 & 0xFFFFu);                   // tile_dim1; tile_dim2=0
  g1[5] = (int)(stride0_elems & 0xFFFFFFFFull);       // dim0_stride[31:0]
  g1[6] = (int)((stride0_elems >> 32) & 0xFFFFull);   // dim0_stride[47:32]
  g1[7] = 0;
  v4i z4 = {0, 0, 0, 0};                              // groups 2/3 unused (2D)
  v8i z8 = {0, 0, 0, 0, 0, 0, 0, 0};
  __builtin_amdgcn_tensor_load_to_lds(g0, g1, z4, z4, z8, 0);
}
#endif

// ---------------------------------------------------------------------------
// RMSNorm: one block per row, D = 1024, 256 threads x float4.
// ---------------------------------------------------------------------------
__global__ __launch_bounds__(256) void rmsnorm_kernel(
    const float* __restrict__ x, const float* __restrict__ g,
    float* __restrict__ out) {
  const int row = blockIdx.x;
  const int tid = threadIdx.x;
  const float4 xv = ((const float4*)(x + (size_t)row * cD))[tid];
  float ss = xv.x * xv.x + xv.y * xv.y + xv.z * xv.z + xv.w * xv.w;
  __shared__ float red[256];
  red[tid] = ss;
  __syncthreads();
  for (int s = 128; s > 0; s >>= 1) {
    if (tid < s) red[tid] += red[tid + s];
    __syncthreads();
  }
  const float inv = rsqrtf(red[0] / (float)cD + 1e-8f);
  const float4 gv = ((const float4*)g)[tid];
  float4 ov;
  ov.x = gv.x * xv.x * inv;
  ov.y = gv.y * xv.y * inv;
  ov.z = gv.z * xv.z * inv;
  ov.w = gv.w * xv.w * inv;
  ((float4*)(out + (size_t)row * cD))[tid] = ov;
}

// ---------------------------------------------------------------------------
// bf16-WMMA GEMM, fully templated: C[M,N] = act(A @ W + bias) (+residual)
// Block tile 128x128, K-step 32, 8 waves (2x4), 64x32 per wave.
// MODE: 0 = fp32 linear (+residual), 1 = bf16 linear,
//       2 = bf16 scatter [B,H,S,dh], 3 = bf16 scatter [B,H,dh,S]
// ---------------------------------------------------------------------------
#define BM 128
#define BN 128
#define BK 32

template <int M, int N, int K, int MODE, bool A_BF16, bool GELU, bool RESID>
__global__ __launch_bounds__(256) void gemm_bf16_kernel(
    const void* __restrict__ Ain, const float* __restrict__ W,
    const float* __restrict__ bias, const float* __restrict__ residual,
    void* __restrict__ out) {
  __shared__ bf16_t As[BM][BK];   // [m][k]
  __shared__ bf16_t Bs[BN][BK];   // [n][k] (transposed on stage-in)

  const int tid   = threadIdx.x;
  const int wave  = tid >> 5;
  const int lane  = tid & 31;
  const int l15   = lane & 15;
  const int khalf = lane >> 4;
  const int waveM = wave >> 2;
  const int waveN = wave & 3;
  const int tileM0 = blockIdx.y * BM;
  const int tileN0 = blockIdx.x * BN;

  v8f acc[4][2] = {};

  for (int k0 = 0; k0 < K; k0 += BK) {
    // Branch-free L2 prefetch of the next weight K-slab.
    {
      size_t knext = (size_t)((k0 + BK < K) ? (k0 + BK) : 0);
      __builtin_prefetch(W + knext * N + tileN0 + lane * 4, 0, 1);
    }

    // ---- stage A ----
    if constexpr (!A_BF16) {
      const float* Af = (const float*)Ain;
      #pragma unroll
      for (int it = 0; it < 4; ++it) {
        int idx = it * 256 + tid;
        int r = idx >> 3, c4 = (idx & 7) * 4;
        float4 f = *(const float4*)(Af + (size_t)(tileM0 + r) * K + k0 + c4);
        v4bf p;
        p.x = (bf16_t)f.x; p.y = (bf16_t)f.y;
        p.z = (bf16_t)f.z; p.w = (bf16_t)f.w;
        *(v4bf*)&As[r][c4] = p;                     // ds_store_b64
      }
    } else {
      const bf16_t* Ab = (const bf16_t*)Ain;
      #pragma unroll
      for (int it = 0; it < 2; ++it) {
        int idx = it * 256 + tid;
        int r = idx >> 2, c8 = (idx & 3) * 8;
        *(v8bf*)&As[r][c8] =
            *(const v8bf*)(Ab + (size_t)(tileM0 + r) * K + k0 + c8);
      }
    }
    // ---- stage W transposed, k-pairs packed into b32 stores ----
    #pragma unroll
    for (int it = 0; it < 2; ++it) {
      int idx = it * 256 + tid;          // 0..511
      int k2 = idx >> 5;                 // k-pair 0..15
      int n0 = (idx & 31) * 4;           // n 0..124
      float4 fa = *(const float4*)(W + (size_t)(k0 + k2 * 2) * N + tileN0 + n0);
      float4 fb = *(const float4*)(W + (size_t)(k0 + k2 * 2 + 1) * N + tileN0 + n0);
      v2bf p0; p0.x = (bf16_t)fa.x; p0.y = (bf16_t)fb.x;
      v2bf p1; p1.x = (bf16_t)fa.y; p1.y = (bf16_t)fb.y;
      v2bf p2; p2.x = (bf16_t)fa.z; p2.y = (bf16_t)fb.z;
      v2bf p3; p3.x = (bf16_t)fa.w; p3.y = (bf16_t)fb.w;
      *(v2bf*)&Bs[n0 + 0][k2 * 2] = p0;             // ds_store_b32
      *(v2bf*)&Bs[n0 + 1][k2 * 2] = p1;
      *(v2bf*)&Bs[n0 + 2][k2 * 2] = p2;
      *(v2bf*)&Bs[n0 + 3][k2 * 2] = p3;
    }
    __syncthreads();

    // ---- fragments + WMMA ----
    V16 afr[4], bfr[2];
    #pragma unroll
    for (int m = 0; m < 4; ++m) {
      int row = waveM * 64 + m * 16 + l15;
      afr[m].h[0] = *(const v8bf*)&As[row][khalf * 8];
      afr[m].h[1] = *(const v8bf*)&As[row][16 + khalf * 8];
    }
    #pragma unroll
    for (int n = 0; n < 2; ++n) {
      int col = waveN * 32 + n * 16 + l15;
      bfr[n].h[0] = *(const v8bf*)&Bs[col][khalf * 8];
      bfr[n].h[1] = *(const v8bf*)&Bs[col][16 + khalf * 8];
    }
    #pragma unroll
    for (int m = 0; m < 4; ++m)
      #pragma unroll
      for (int n = 0; n < 2; ++n)
        acc[m][n] = wmma_bf16(afr[m].v, bfr[n].v, acc[m][n]);
    __syncthreads();
  }

  // ---- epilogue ----
  #pragma unroll
  for (int m = 0; m < 4; ++m) {
    #pragma unroll
    for (int n = 0; n < 2; ++n) {
      #pragma unroll
      for (int r = 0; r < 8; ++r) {
        int grow = tileM0 + waveM * 64 + m * 16 + khalf * 8 + r;
        int gcol = tileN0 + waveN * 32 + n * 16 + l15;
        float v = acc[m][n][r] + bias[gcol];
        if constexpr (GELU) v = gelu_exact(v);
        if constexpr (MODE == 0) {
          if constexpr (RESID) v += residual[(size_t)grow * N + gcol];
          ((float*)out)[(size_t)grow * N + gcol] = v;
        } else if constexpr (MODE == 1) {
          ((bf16_t*)out)[(size_t)grow * N + gcol] = (bf16_t)v;
        } else {
          int b = grow / cS, s = grow - b * cS;
          int hh = gcol / cDH, d = gcol - hh * cDH;
          if constexpr (MODE == 2)  // [B,H,S,dh]
            ((bf16_t*)out)[(((size_t)b * cH + hh) * cS + s) * cDH + d] = (bf16_t)v;
          else                      // [B,H,dh,S]
            ((bf16_t*)out)[(((size_t)b * cH + hh) * cDH + d) * cS + s] = (bf16_t)v;
        }
      }
    }
  }
}

// ---------------------------------------------------------------------------
// Fused flash attention. dh = 64, 8 waves, 16 q-rows/wave, 128-key tiles.
// K/V tiles staged to LDS by the Tensor Data Mover when available.
// q,k: bf16 [B,H,S,dh] ; v: bf16 [B,H,dh,S] ; out: fp32 [B,S,D]
// ---------------------------------------------------------------------------
__global__ __launch_bounds__(256) void attn_kernel(
    const bf16_t* __restrict__ q, const bf16_t* __restrict__ k,
    const bf16_t* __restrict__ v, const float* __restrict__ rel_bias,
    const unsigned char* __restrict__ pmask, float* __restrict__ out) {
  __shared__ bf16_t Ks[128][64];      // [key][d]
  __shared__ bf16_t Vs[64][128];      // [d][key]
  __shared__ bf16_t Ps[8][16][128];   // per-wave P tile

  const int qt = blockIdx.x;
  const int bh = blockIdx.y;
  const int b  = bh / cH;
  const int hh = bh - b * cH;
  const int tid   = threadIdx.x;
  const int wave  = tid >> 5;
  const int lane  = tid & 31;
  const int l15   = lane & 15;
  const int khalf = lane >> 4;
  const int qbase = qt * 128 + wave * 16;

  V16 qf[2];
  {
    const bf16_t* qp = q + ((size_t)bh * cS + qbase + l15) * cDH;
    #pragma unroll
    for (int kk = 0; kk < 2; ++kk) {
      qf[kk].h[0] = *(const v8bf*)(qp + kk * 32 + khalf * 8);
      qf[kk].h[1] = *(const v8bf*)(qp + kk * 32 + 16 + khalf * 8);
    }
  }

  float m_r[8], l_r[8];
  #pragma unroll
  for (int r = 0; r < 8; ++r) { m_r[r] = -1e30f; l_r[r] = 0.0f; }
  v8f o_[4] = {};
  const float sc = 0.125f;  // 1/sqrt(64)

#if USE_TDM
  const unsigned ksOff = (unsigned)(unsigned long long)(void*)&Ks[0][0];
  const unsigned vsOff = (unsigned)(unsigned long long)(void*)&Vs[0][0];
#endif

  for (int kt0 = 0; kt0 < cS; kt0 += 128) {
    __syncthreads();  // previous tile's consumers done
#if USE_TDM
    if (wave == 0) {
      // K tile: 128 rows x 64 bf16, row stride 64 elements.
      tdm_load_2d_bf16(ksOff, k + ((size_t)bh * cS + kt0) * cDH,
                       /*tensor_d0=*/cDH, /*tensor_d1=*/cS,
                       /*tile_d0=*/cDH, /*tile_d1=*/128, /*stride0=*/cDH);
      // V^T tile: 64 rows x 128 bf16, row stride S elements.
      tdm_load_2d_bf16(vsOff, v + ((size_t)bh * cDH) * cS + kt0,
                       /*tensor_d0=*/cS, /*tensor_d1=*/cDH,
                       /*tile_d0=*/128, /*tile_d1=*/cDH, /*stride0=*/cS);
      __builtin_amdgcn_s_wait_tensorcnt(0);
    }
#else
    #pragma unroll
    for (int it = 0; it < 4; ++it) {
      int idx = it * 256 + tid;
      int r = idx >> 3, c8 = (idx & 7) * 8;
      *(v8bf*)&Ks[r][c8] =
          *(const v8bf*)(k + ((size_t)bh * cS + kt0 + r) * cDH + c8);
    }
    #pragma unroll
    for (int it = 0; it < 4; ++it) {
      int idx = it * 256 + tid;
      int r = idx >> 4, c8 = (idx & 15) * 8;
      *(v8bf*)&Vs[r][c8] =
          *(const v8bf*)(v + ((size_t)bh * cDH + r) * cS + kt0 + c8);
    }
#endif
    __syncthreads();

    // ---- scores ----
    v8f s_[8];
    #pragma unroll
    for (int ct = 0; ct < 8; ++ct) {
      const int key = ct * 16 + l15;
      V16 kb0, kb1;
      kb0.h[0] = *(const v8bf*)&Ks[key][khalf * 8];
      kb0.h[1] = *(const v8bf*)&Ks[key][16 + khalf * 8];
      kb1.h[0] = *(const v8bf*)&Ks[key][32 + khalf * 8];
      kb1.h[1] = *(const v8bf*)&Ks[key][32 + 16 + khalf * 8];
      v8f a = {};
      a = wmma_bf16(qf[0].v, kb0.v, a);
      a = wmma_bf16(qf[1].v, kb1.v, a);
      const int kcol = kt0 + key;
      const bool valid = pmask[(size_t)b * cS + kcol] != 0;
      #pragma unroll
      for (int r = 0; r < 8; ++r) {
        int qrow = qbase + khalf * 8 + r;
        float bv = rel_bias[((size_t)hh * cS + qrow) * cS + kcol];
        float val = a[r] * sc + bv;
        s_[ct][r] = valid ? val : -1e30f;
      }
    }

    // ---- online softmax ----
    float osc[8];
    #pragma unroll
    for (int r = 0; r < 8; ++r) {
      float tmax = -1e30f;
      #pragma unroll
      for (int ct = 0; ct < 8; ++ct) tmax = fmaxf(tmax, s_[ct][r]);
      #pragma unroll
      for (int off = 1; off < 16; off <<= 1)
        tmax = fmaxf(tmax, __shfl_xor(tmax, off, 32));
      float mnew = fmaxf(m_r[r], tmax);
      osc[r] = __expf(m_r[r] - mnew);
      m_r[r] = mnew;
      l_r[r] *= osc[r];
    }
    #pragma unroll
    for (int dt = 0; dt < 4; ++dt)
      #pragma unroll
      for (int r = 0; r < 8; ++r) o_[dt][r] *= osc[r];

    float ps[8] = {};
    #pragma unroll
    for (int ct = 0; ct < 8; ++ct) {
      #pragma unroll
      for (int r = 0; r < 8; ++r) {
        float p = __expf(s_[ct][r] - m_r[r]);
        ps[r] += p;
        Ps[wave][khalf * 8 + r][ct * 16 + l15] = (bf16_t)p;
      }
    }
    #pragma unroll
    for (int r = 0; r < 8; ++r) {
      float t = ps[r];
      #pragma unroll
      for (int off = 1; off < 16; off <<= 1) t += __shfl_xor(t, off, 32);
      l_r[r] += t;
    }

    // ---- O += P @ V ----
    #pragma unroll
    for (int kk = 0; kk < 4; ++kk) {
      V16 pf;
      pf.h[0] = *(const v8bf*)&Ps[wave][l15][kk * 32 + khalf * 8];
      pf.h[1] = *(const v8bf*)&Ps[wave][l15][kk * 32 + 16 + khalf * 8];
      #pragma unroll
      for (int dt = 0; dt < 4; ++dt) {
        const int dcol = dt * 16 + l15;
        V16 vf;
        vf.h[0] = *(const v8bf*)&Vs[dcol][kk * 32 + khalf * 8];
        vf.h[1] = *(const v8bf*)&Vs[dcol][kk * 32 + 16 + khalf * 8];
        o_[dt] = wmma_bf16(pf.v, vf.v, o_[dt]);
      }
    }
  }

  // ---- normalize and write [B,S,D] ----
  #pragma unroll
  for (int r = 0; r < 8; ++r) {
    float inv = l_r[r] > 0.0f ? 1.0f / l_r[r] : 0.0f;
    int qrow = qbase + khalf * 8 + r;
    #pragma unroll
    for (int dt = 0; dt < 4; ++dt) {
      int d = dt * 16 + l15;
      out[((size_t)b * cS + qrow) * cD + hh * cDH + d] = o_[dt][r] * inv;
    }
  }
}

// ---------------------------------------------------------------------------
// Host-side orchestration.
// ---------------------------------------------------------------------------
extern "C" void kernel_launch(void* const* d_in, const int* in_sizes, int n_in,
                              void* d_out, int out_size, void* d_ws, size_t ws_size,
                              hipStream_t stream) {
  const float* x        = (const float*)d_in[0];
  const unsigned char* pmask = (const unsigned char*)d_in[1];
  const float* rel_bias = (const float*)d_in[2];
  const float* Wq = (const float*)d_in[3];
  const float* bq = (const float*)d_in[4];
  const float* Wk = (const float*)d_in[5];
  const float* bk = (const float*)d_in[6];
  const float* Wv = (const float*)d_in[7];
  const float* bv = (const float*)d_in[8];
  const float* Wo = (const float*)d_in[9];
  const float* bo = (const float*)d_in[10];
  const float* g1 = (const float*)d_in[11];
  const float* w1 = (const float*)d_in[12];
  const float* b1 = (const float*)d_in[13];
  const float* w2 = (const float*)d_in[14];
  const float* b2 = (const float*)d_in[15];
  const float* g2 = (const float*)d_in[16];
  float* outF = (float*)d_out;

  char* ws = (char*)d_ws;
  size_t off = 0;
  float*  h    = (float*)(ws + off);  off += (size_t)cBS * cD * 4;  // 16 MB
  bf16_t* qb   = (bf16_t*)(ws + off); off += (size_t)cBS * cD * 2;  //  8 MB
  bf16_t* kb   = (bf16_t*)(ws + off); off += (size_t)cBS * cD * 2;  //  8 MB
  bf16_t* vb   = (bf16_t*)(ws + off); off += (size_t)cBS * cD * 2;  //  8 MB
  float*  attn = (float*)(ws + off);  off += (size_t)cBS * cD * 4;  // 16 MB
  bf16_t* ff1  = (bf16_t*)(ws + off); off += (size_t)cBS * cF * 2;  // 32 MB

  const dim3 blk(256);
  const dim3 gridProj(cD / BN, cBS / BM);
  const dim3 gridFF1(cF / BN, cBS / BM);
  const dim3 gridAttn(cS / 128, cB * cH);

  // 1) h = rmsnorm(x, g1)
  rmsnorm_kernel<<<cBS, blk, 0, stream>>>(x, g1, h);

  // 2) Q/K/V projections -> bf16, head-major layouts
  gemm_bf16_kernel<cBS, cD, cD, 2, false, false, false>
      <<<gridProj, blk, 0, stream>>>(h, Wq, bq, nullptr, qb);
  gemm_bf16_kernel<cBS, cD, cD, 2, false, false, false>
      <<<gridProj, blk, 0, stream>>>(h, Wk, bk, nullptr, kb);
  gemm_bf16_kernel<cBS, cD, cD, 3, false, false, false>
      <<<gridProj, blk, 0, stream>>>(h, Wv, bv, nullptr, vb);

  // 3) fused attention -> attn [B,S,D] fp32
  attn_kernel<<<gridAttn, blk, 0, stream>>>(qb, kb, vb, rel_bias, pmask, attn);

  // 4) x1 = x + attn @ Wo + bo -> d_out
  gemm_bf16_kernel<cBS, cD, cD, 0, false, false, true>
      <<<gridProj, blk, 0, stream>>>(attn, Wo, bo, x, outF);

  // 5) h = rmsnorm(x1, g2)
  rmsnorm_kernel<<<cBS, blk, 0, stream>>>(outF, g2, h);

  // 6) ff1 = gelu(h @ w1 + b1) -> bf16
  gemm_bf16_kernel<cBS, cF, cD, 1, false, true, false>
      <<<gridFF1, blk, 0, stream>>>(h, w1, b1, nullptr, ff1);

  // 7) d_out = x1 + ff1 @ w2 + b2
  gemm_bf16_kernel<cBS, cD, cF, 0, true, false, true>
      <<<gridProj, blk, 0, stream>>>(ff1, w2, b2, outF, outF);

  (void)in_sizes; (void)n_in; (void)out_size; (void)ws_size; (void)off;
}
